// MultiHeadAttention_74594991997050
// MI455X (gfx1250) — matrix operand order repaired and checked by reading, not compiled, not run on hardware
//
#include <hip/hip_runtime.h>
#include <hip/hip_bf16.h>
#include <math.h>

// ---------------------------------------------------------------------------
// CDNA5 (gfx1250) Shaw relative-position MHA, bf16 WMMA + TDM throughout.
// ---------------------------------------------------------------------------

typedef __attribute__((ext_vector_type(16))) __bf16 v16bf;
typedef __attribute__((ext_vector_type(8)))  float  v8f;
typedef __attribute__((ext_vector_type(4)))  unsigned int v4u;
typedef __attribute__((ext_vector_type(8)))  int    v8i_t;
typedef __attribute__((ext_vector_type(4)))  int    v4i_t;

#define DEV static __device__ __forceinline__

#define BB  8
#define SS  1024
#define DD  1024
#define HH  16
#define DKK 64
#define NR  65          // 2*MAX_REL+1

#if __has_builtin(__builtin_amdgcn_tensor_load_to_lds) && \
    __has_builtin(__builtin_amdgcn_s_wait_tensorcnt)
#define HAS_TDM 1
#if defined(__has_include)
#if __has_include(<hip/amd_detail/amd_gfx1250_TDM.h>)
#define TDM_ARGS6 1
#else
#define TDM_ARGS6 0
#endif
#else
#define TDM_ARGS6 0
#endif
#else
#define HAS_TDM 0
#endif

// ---- bf16 helpers (bit-exact RNE convert) ----------------------------------
DEV unsigned short f2bf_us(float f) {
  unsigned int u = __float_as_uint(f);
  u += 0x7FFFu + ((u >> 16) & 1u);
  return (unsigned short)(u >> 16);
}
DEV float bf2f(unsigned short u) { return __uint_as_float(((unsigned int)u) << 16); }
DEV __bf16 us2bf(unsigned short u) { __bf16 b; __builtin_memcpy(&b, &u, 2); return b; }
DEV __bf16 f2bf(float f) { return us2bf(f2bf_us(f)); }

// ---- WMMA fragment loaders (ISA 7.12.2 16-bit layouts, wave32) -------------
// A fragment 16x32 (MxK), row-major source: lane&15 = M row; contiguous ->
// compiler emits 2x ds_load_b128 per fragment.
DEV v16bf load_a_frag(const unsigned short* t, int stride) {
  int lane = threadIdx.x & 31;
  const unsigned short* p = t + (lane & 15) * stride + ((lane >> 4) << 3);
  v16bf a;
#pragma unroll
  for (int i = 0; i < 8; ++i) { a[i] = us2bf(p[i]); a[8 + i] = us2bf(p[16 + i]); }
  return a;
}
// B fragment 32x16 (KxN) from K-major source: B[k][n] = src[k*stride + n].
// lane holds column n = lane&15; k = 8*(lane>>4)+i and 16+8*(lane>>4)+i.
DEV v16bf load_b_frag_rm(const unsigned short* t, int stride) {
  int lane = threadIdx.x & 31;
  const unsigned short* p = t + (lane & 15) + ((lane >> 4) << 3) * stride;
  v16bf b;
#pragma unroll
  for (int i = 0; i < 8; ++i) { b[i] = us2bf(p[i * stride]); b[8 + i] = us2bf(p[(16 + i) * stride]); }
  return b;
}

DEV v8f wmma_bf16(v16bf a, v16bf b, v8f c) {
  return __builtin_amdgcn_wmma_f32_16x16x32_bf16(false, a, false, b, (short)0, c,
                                                 false, false);
}

// ---- Tensor Data Mover: 2-D tile, 2-byte elements, padded LDS rows ---------
// tile_d0 elems/row, tile_d1 rows, global row stride in elems; pad codes per
// D# spec (interval: 0=2DW..4=32DW; amount: code+1 DWORDs).
DEV void tdm_load_2d(unsigned lds_off, const void* gptr, int tile_d0, int tile_d1,
                     long long stride_elems, int pad_int_code, int pad_amt_code) {
#if HAS_TDM
  unsigned long long ga = (unsigned long long)gptr;
  v4u g0;
  g0[0] = 1u;                                            // count=1, user mode
  g0[1] = lds_off;                                       // lds_addr (bytes)
  g0[2] = (unsigned)(ga & 0xffffffffu);                  // global_addr lo
  g0[3] = (unsigned)((ga >> 32) & 0x01ffffffu) | (2u << 30);  // addr hi | type=2
  v8i_t g1;
  g1[0] = (1 << 16) | (1 << 20) | (pad_int_code << 22) | (pad_amt_code << 25);
  g1[1] = (tile_d0 & 0xffff) << 16;                      // tensor_dim0 lo16
  g1[2] = ((tile_d0 >> 16) & 0xffff) | ((tile_d1 & 0xffff) << 16); // dim0 hi | dim1 lo
  g1[3] = ((tile_d1 >> 16) & 0xffff) | ((tile_d0 & 0xffff) << 16); // dim1 hi | tile_dim0
  g1[4] = tile_d1 & 0xffff;                              // tile_dim1 (tile_dim2=0)
  g1[5] = (int)(stride_elems & 0xffffffffll);            // tensor_dim0_stride lo
  g1[6] = (int)((stride_elems >> 32) & 0xffffll);        // stride hi | dim1_stride lo
  g1[7] = 0;
  v4i_t z = {0, 0, 0, 0};
#if TDM_ARGS6
  v8i_t z8 = {0, 0, 0, 0, 0, 0, 0, 0};
  __builtin_amdgcn_tensor_load_to_lds(g0, g1, z, z, z8, 0);
#else
  __builtin_amdgcn_tensor_load_to_lds(g0, g1, z, z, 0);
#endif
#endif
}

// ---------------------------------------------------------------------------
// Kernel 1: f32 -> bf16 pack (grid-stride)
// ---------------------------------------------------------------------------
__global__ void cvt_bf16_kernel(const float* __restrict__ in,
                                unsigned short* __restrict__ out, int n) {
  for (int i = blockIdx.x * blockDim.x + threadIdx.x; i < n;
       i += gridDim.x * blockDim.x)
    out[i] = f2bf_us(in[i]);
}

// ---------------------------------------------------------------------------
// Kernel 2: bf16 GEMM  out[m][n] = scale * sum_k A[m][k]*W[n][k]  (x @ W.T)
// 128(M) x 64(N) workgroup tile, K panels of 32.  A panel DMA'd by the TDM
// (padded rows), W panel staged transposed so B-fragments are contiguous.
// ---------------------------------------------------------------------------
__global__ __launch_bounds__(256) void gemm_bf16_kernel(
    const unsigned short* __restrict__ A, const unsigned short* __restrict__ W,
    unsigned short* __restrict__ outB, float* __restrict__ outF,
    const float* __restrict__ bias, float scale, int writeF) {
  __shared__ __align__(16) unsigned short sA[128 * 40];   // row-major, pad 8
  __shared__ __align__(16) unsigned short sBt[32 * 72];   // k-major (transposed)

  const int lane = threadIdx.x & 31, wv = threadIdx.x >> 5;
  const int laneHi = lane >> 4, nIdx = lane & 15;
  const int mi = (wv & 3) * 32, ni = (wv >> 2) * 32;
  const int m0 = blockIdx.x * 128, n0 = blockIdx.y * 64;

  v8f acc[2][2] = {{}, {}};

  for (int kb = 0; kb < DD; kb += 32) {
    __syncthreads();
#if HAS_TDM
    if (threadIdx.x < 32) {   // wave 0 drives the DMA of the A panel
      tdm_load_2d((unsigned)(unsigned long long)(const void*)sA,
                  A + (size_t)m0 * DD + kb, 32, 128, DD, /*16DW*/ 3, /*4DW*/ 3);
      __builtin_amdgcn_s_wait_tensorcnt(0);
    }
#else
#pragma unroll
    for (int c = threadIdx.x; c < 512; c += 256) {
      int r = c >> 2, p = c & 3;
      *(uint2*)&sA[r * 40 + p * 8] =
          *(const uint2*)(A + (size_t)(m0 + r) * DD + kb + p * 8);
    }
#endif
    // stage W panel transposed: sBt[k][n] = W[n0+n][kb+k]
#pragma unroll
    for (int c = threadIdx.x; c < 512; c += 256) {
      int n = c >> 3, kq = c & 7;
      union { uint2 u; unsigned short s[4]; } cv;
      cv.u = *(const uint2*)(W + (size_t)(n0 + n) * DD + kb + kq * 4);
#pragma unroll
      for (int e = 0; e < 4; ++e) sBt[(kq * 4 + e) * 72 + n] = cv.s[e];
    }
    if (kb + 32 < DD)
      __builtin_prefetch(W + (size_t)(n0 + (threadIdx.x >> 2)) * DD + kb + 32, 0, 1);
    __syncthreads();

    v16bf a0 = load_a_frag(sA + (mi)*40, 40);
    v16bf a1 = load_a_frag(sA + (mi + 16) * 40, 40);
    v16bf b0 = load_b_frag_rm(sBt + ni, 72);
    v16bf b1 = load_b_frag_rm(sBt + ni + 16, 72);
    acc[0][0] = wmma_bf16(a0, b0, acc[0][0]);
    acc[0][1] = wmma_bf16(a0, b1, acc[0][1]);
    acc[1][0] = wmma_bf16(a1, b0, acc[1][0]);
    acc[1][1] = wmma_bf16(a1, b1, acc[1][1]);
  }

#pragma unroll
  for (int ti = 0; ti < 2; ++ti)
#pragma unroll
    for (int tj = 0; tj < 2; ++tj)
#pragma unroll
      for (int v = 0; v < 8; ++v) {
        int row = m0 + mi + ti * 16 + v + 8 * laneHi;
        int col = n0 + ni + tj * 16 + nIdx;
        float x = acc[ti][tj][v] * scale;
        if (writeF)
          outF[(size_t)row * DD + col] = x + bias[col];
        else
          outB[(size_t)row * DD + col] = f2bf_us(x);
      }
}

// ---------------------------------------------------------------------------
// Kernel 3: Qrel[bs, h, r] = sum_d Qb[bs, h*64+d] * rel_k_emb[r, d]
// ---------------------------------------------------------------------------
__global__ void qrel_kernel(const unsigned short* __restrict__ Qb,
                            const float* __restrict__ relk,
                            float* __restrict__ Qrel, int total) {
  int idx = blockIdx.x * blockDim.x + threadIdx.x;
  if (idx >= total) return;
  int r = idx % NR;
  int t = idx / NR;
  int h = t % HH;
  size_t bs = (size_t)(t / HH);
  const unsigned short* q = Qb + bs * DD + h * DKK;
  const float* rk = relk + r * DKK;
  float acc = 0.f;
#pragma unroll
  for (int d = 0; d < DKK; ++d) acc += bf2f(q[d]) * rk[d];
  Qrel[bs * (HH * NR) + h * NR + r] = acc;
}

// ---------------------------------------------------------------------------
// Kernel 4: flash-style attention with Shaw relative terms (scores^T scheme).
// Grid (S/64, H, B), 128 threads (4 waves), each wave owns 16 query rows.
//   scores^T = K-tile(A,row-major) x Q^T(B, built once)  -> C: M=j, N=i
//   -> per-lane softmax rows (i = lane&15), one shfl_xor(16) per reduction
//   -> prob C-frag == P A-frag layout (zero-shuffle repack)
//   -> ctx = P(A) x V-tile(B, row-major contiguous)
// K/V tiles DMA'd into padded LDS rows by the TDM; bucket weights w[i][r]
// accumulated with ds_add_f32; epilogue adds w @ rel_v_emb and normalizes.
// ---------------------------------------------------------------------------
__global__ __launch_bounds__(128) void attn_kernel(
    const unsigned short* __restrict__ Qb, const unsigned short* __restrict__ Kb,
    const unsigned short* __restrict__ Vb, const float* __restrict__ Qrel,
    const float* __restrict__ relv, unsigned short* __restrict__ ctxb) {
  __shared__ __align__(16) unsigned short sQ[64 * 72];
  __shared__ __align__(16) unsigned short sK[32 * 72];
  __shared__ __align__(16) unsigned short sV[32 * 72];
  __shared__ __align__(16) unsigned short sRelV[NR * DKK];
  __shared__ float sQrel[64][66];
  __shared__ float sW[4][16][66];
  __shared__ float sL[4][16];

  const int b = blockIdx.z, h = blockIdx.y, ib = blockIdx.x;
  const int tid = threadIdx.x, lane = tid & 31, wv = tid >> 5;
  const int laneHi = lane >> 4, nIdx = lane & 15;
  const size_t rowBase = (size_t)b * SS + (size_t)ib * 64;

  // ---- prologue staging ----
  for (int i = tid; i < NR * DKK; i += 128) sRelV[i] = f2bf_us(relv[i]);
#pragma unroll
  for (int c = tid; c < 64 * 8; c += 128) {
    int r = c >> 3, p = c & 7;
    *(uint2*)&sQ[r * 72 + p * 8] =
        *(const uint2*)(Qb + (rowBase + r) * DD + h * DKK + p * 8);
  }
  for (int c = tid; c < 64 * NR; c += 128) {
    int r = c / NR, e = c % NR;
    sQrel[r][e] = Qrel[(rowBase + r) * (HH * NR) + h * NR + e];
  }
  for (int c = tid; c < 4 * 16 * 66; c += 128) (&sW[0][0][0])[c] = 0.f;
  __syncthreads();

  // Q^T B-fragments (built once): B[k=d][n=i] = Q[i][d]
  v16bf bq0, bq1;
  {
    int n = lane & 15, k = (lane >> 4) << 3;
    const unsigned short* q0 = sQ + (wv * 16) * 72;
#pragma unroll
    for (int i = 0; i < 8; ++i) {
      bq0[i] = us2bf(q0[n * 72 + k + i]);
      bq0[8 + i] = us2bf(q0[n * 72 + 16 + k + i]);
      bq1[i] = us2bf(q0[n * 72 + 32 + k + i]);
      bq1[8 + i] = us2bf(q0[n * 72 + 48 + k + i]);
    }
  }

  const int iL = wv * 16 + nIdx;   // row within workgroup
  const int iG = ib * 64 + iL;     // global sequence position
  float m_i = -1e30f, l_i = 0.f;

  // ---- pass 1: online softmax statistics ----
  for (int jb = 0; jb < SS; jb += 32) {
    __syncthreads();
#if HAS_TDM
    if (tid < 32) {
      tdm_load_2d((unsigned)(unsigned long long)(const void*)sK,
                  Kb + ((size_t)b * SS + jb) * DD + h * DKK, 64, 32, DD,
                  /*32DW*/ 4, /*4DW*/ 3);
      __builtin_amdgcn_s_wait_tensorcnt(0);
    }
#else
#pragma unroll
    for (int c = tid; c < 32 * 8; c += 128) {
      int r = c >> 3, p = c & 7;
      *(uint2*)&sK[r * 72 + p * 8] =
          *(const uint2*)(Kb + ((size_t)b * SS + jb + r) * DD + h * DKK + p * 8);
    }
#endif
    __syncthreads();

    float st[2][8];
#pragma unroll
    for (int jt = 0; jt < 2; ++jt) {
      v8f s = {};
      v16bf ak0 = load_a_frag(sK + (jt * 16) * 72 + 0, 72);
      v16bf ak1 = load_a_frag(sK + (jt * 16) * 72 + 32, 72);
      s = wmma_bf16(ak0, bq0, s);
      s = wmma_bf16(ak1, bq1, s);
#pragma unroll
      for (int v = 0; v < 8; ++v) {
        int j = jb + jt * 16 + v + 8 * laneHi;
        int rel = j - iG;
        rel = rel < -32 ? -32 : (rel > 32 ? 32 : rel);
        st[jt][v] = s[v] + sQrel[iL][rel + 32];
      }
    }
    float mx = st[0][0];
#pragma unroll
    for (int jt = 0; jt < 2; ++jt)
#pragma unroll
      for (int v = 0; v < 8; ++v) mx = fmaxf(mx, st[jt][v]);
    mx = fmaxf(mx, __shfl_xor(mx, 16, 32));
    float mnew = fmaxf(m_i, mx);
    float ps = 0.f;
#pragma unroll
    for (int jt = 0; jt < 2; ++jt)
#pragma unroll
      for (int v = 0; v < 8; ++v) ps += __expf(st[jt][v] - mnew);
    ps += __shfl_xor(ps, 16, 32);
    l_i = l_i * __expf(m_i - mnew) + ps;
    m_i = mnew;
  }

  // ---- pass 2: context accumulation ----
  v8f ct[4] = {};
  for (int jb = 0; jb < SS; jb += 32) {
    __syncthreads();
#if HAS_TDM
    if (tid < 32) {
      tdm_load_2d((unsigned)(unsigned long long)(const void*)sK,
                  Kb + ((size_t)b * SS + jb) * DD + h * DKK, 64, 32, DD, 4, 3);
      tdm_load_2d((unsigned)(unsigned long long)(const void*)sV,
                  Vb + ((size_t)b * SS + jb) * DD + h * DKK, 64, 32, DD, 4, 3);
      __builtin_amdgcn_s_wait_tensorcnt(0);
    }
#else
#pragma unroll
    for (int c = tid; c < 512; c += 128) {
      int r = (c >> 3) & 31, p = c & 7;
      size_t g = ((size_t)b * SS + jb + r) * DD + h * DKK + p * 8;
      if (c < 256)
        *(uint2*)&sK[r * 72 + p * 8] = *(const uint2*)(Kb + g);
      else
        *(uint2*)&sV[r * 72 + p * 8] = *(const uint2*)(Vb + g);
    }
#endif
    __syncthreads();

    v16bf ap;   // prob A-fragment: direct repack, no shuffles
#pragma unroll
    for (int jt = 0; jt < 2; ++jt) {
      v8f s = {};
      v16bf ak0 = load_a_frag(sK + (jt * 16) * 72 + 0, 72);
      v16bf ak1 = load_a_frag(sK + (jt * 16) * 72 + 32, 72);
      s = wmma_bf16(ak0, bq0, s);
      s = wmma_bf16(ak1, bq1, s);
#pragma unroll
      for (int v = 0; v < 8; ++v) {
        int j = jb + jt * 16 + v + 8 * laneHi;
        int rel = j - iG;
        rel = rel < -32 ? -32 : (rel > 32 ? 32 : rel);
        float p = __expf(s[v] + sQrel[iL][rel + 32] - m_i);
        atomicAdd(&sW[wv][nIdx][rel + 32], p);   // bucket weights (ds_add_f32)
        ap[jt * 8 + v] = f2bf(p);
      }
    }
#pragma unroll
    for (int t = 0; t < 4; ++t) {
      v16bf bv = load_b_frag_rm(sV + t * 16, 72);
      ct[t] = wmma_bf16(ap, bv, ct[t]);
    }
  }
  __syncthreads();
  sL[wv][nIdx] = l_i;   // both half-wave copies write identical values
  __syncthreads();

  // ---- epilogue: rel-value term + normalize + store ----
  // ctx C-frag: N=d (nIdx within d-tile t), M=i (v + 8*laneHi within wave)
  float accv[4][8];
#pragma unroll
  for (int t = 0; t < 4; ++t)
#pragma unroll
    for (int v = 0; v < 8; ++v) accv[t][v] = ct[t][v];

  for (int r = 0; r < NR; ++r) {
    float w8[8];
#pragma unroll
    for (int v = 0; v < 8; ++v) w8[v] = sW[wv][v + 8 * laneHi][r];
#pragma unroll
    for (int t = 0; t < 4; ++t) {
      float rv = bf2f(sRelV[r * DKK + t * 16 + nIdx]);
#pragma unroll
      for (int v = 0; v < 8; ++v) accv[t][v] += w8[v] * rv;
    }
  }
  float invl8[8];
#pragma unroll
  for (int v = 0; v < 8; ++v) invl8[v] = 1.f / sL[wv][v + 8 * laneHi];
#pragma unroll
  for (int t = 0; t < 4; ++t)
#pragma unroll
    for (int v = 0; v < 8; ++v)
      ctxb[(rowBase + wv * 16 + v + 8 * laneHi) * DD + h * DKK + t * 16 + nIdx] =
          f2bf_us(accv[t][v] * invl8[v]);
}

// ---------------------------------------------------------------------------
// Host-side launch sequencing (all on `stream`, graph-capture safe).
// ---------------------------------------------------------------------------
extern "C" void kernel_launch(void* const* d_in, const int* in_sizes, int n_in,
                              void* d_out, int out_size, void* d_ws, size_t ws_size,
                              hipStream_t stream) {
  const float* x    = (const float*)d_in[0];
  const float* Wq   = (const float*)d_in[1];
  const float* Wk   = (const float*)d_in[2];
  const float* Wv   = (const float*)d_in[3];
  const float* Wo   = (const float*)d_in[4];
  const float* b_o  = (const float*)d_in[5];
  const float* relk = (const float*)d_in[6];
  const float* relv = (const float*)d_in[7];
  float* out = (float*)d_out;

  const size_t SZ_X = (size_t)BB * SS * DD * 2;  // bf16 activations
  const size_t SZ_W = (size_t)DD * DD * 2;       // bf16 weight
  char* ws = (char*)d_ws;
  unsigned short* xb   = (unsigned short*)(ws);
  unsigned short* wqb  = (unsigned short*)(ws + SZ_X);
  unsigned short* wkb  = (unsigned short*)(ws + SZ_X + SZ_W);
  unsigned short* wvb  = (unsigned short*)(ws + SZ_X + 2 * SZ_W);
  unsigned short* wob  = (unsigned short*)(ws + SZ_X + 3 * SZ_W);
  unsigned short* qb   = (unsigned short*)(ws + SZ_X + 4 * SZ_W);
  unsigned short* kb   = (unsigned short*)(ws + 2 * SZ_X + 4 * SZ_W);
  unsigned short* vb   = (unsigned short*)(ws + 3 * SZ_X + 4 * SZ_W);
  unsigned short* ctxb = (unsigned short*)(ws + 4 * SZ_X + 4 * SZ_W);
  float* qrel          = (float*)(ws + 5 * SZ_X + 4 * SZ_W);

  const float inv_sqrt_dk = 0.125f;  // 1/sqrt(64)

  // 1) pack to bf16
  cvt_bf16_kernel<<<4096, 256, 0, stream>>>(x, xb, BB * SS * DD);
  cvt_bf16_kernel<<<1024, 256, 0, stream>>>(Wq, wqb, DD * DD);
  cvt_bf16_kernel<<<1024, 256, 0, stream>>>(Wk, wkb, DD * DD);
  cvt_bf16_kernel<<<1024, 256, 0, stream>>>(Wv, wvb, DD * DD);
  cvt_bf16_kernel<<<1024, 256, 0, stream>>>(Wo, wob, DD * DD);

  // 2) projections (Q pre-scaled by 1/sqrt(dk))
  dim3 ggrid(BB * SS / 128, DD / 64);
  gemm_bf16_kernel<<<ggrid, 256, 0, stream>>>(xb, wqb, qb, nullptr, nullptr,
                                              inv_sqrt_dk, 0);
  gemm_bf16_kernel<<<ggrid, 256, 0, stream>>>(xb, wkb, kb, nullptr, nullptr, 1.f, 0);
  gemm_bf16_kernel<<<ggrid, 256, 0, stream>>>(xb, wvb, vb, nullptr, nullptr, 1.f, 0);

  // 3) Qrel = Q @ rel_k_emb^T
  {
    int total = BB * SS * HH * NR;
    qrel_kernel<<<(total + 255) / 256, 256, 0, stream>>>(qb, relk, qrel, total);
  }

  // 4) attention (flash two-pass, Shaw rel terms)
  attn_kernel<<<dim3(SS / 64, HH, BB), 128, 0, stream>>>(qb, kb, vb, qrel, relv,
                                                         ctxb);

  // 5) output projection + bias (f32 out)
  gemm_bf16_kernel<<<ggrid, 256, 0, stream>>>(ctxb, wob, nullptr, out, b_o, 1.f, 1);
}
